// VectorQuantizer_53369263620694
// MI455X (gfx1250) — matrix operand order repaired and checked by reading, not compiled, not run on hardware
//
#include <hip/hip_runtime.h>
#include <stdint.h>

// ---------------------------------------------------------------------------
// VQ nearest-codebook kernel for gfx1250 (MI455X), wave32 + WMMA bf16.
//   scores:  s[b,k] = ||e_k||^2 - 2 * z_b . e_k        (||z||^2 is argmin-invariant)
//   GEMM:    z (131072x256) @ codebook^T (256x512) via v_wmma_f32_16x16x32_bf16
//            with hi/lo bf16 split for ~fp32 accuracy (3 WMMAs per K=32 chunk).
//   Round 3: B tiles staged through LDS (double-buffered, bank-conflict-free),
//            shared by 8 waves per block -> 8x less L2 traffic.
// ---------------------------------------------------------------------------

typedef __attribute__((ext_vector_type(16))) __bf16 v16bf;
typedef __attribute__((ext_vector_type(8)))  float  v8f;

#define NUM_EMB 512
#define D_EMB   256
#define NROWS   131072
#define WAVES_PER_BLK 8
#define ROWS_PER_BLK  (WAVES_PER_BLK * 16)       // 128
#define SCORE_BLOCKS  (NROWS / ROWS_PER_BLK)     // 1024
#define LOSS_BLOCKS   (NROWS / 64)               // 2048
#define BROW 264    // padded LDS row stride in ushorts (132 dwords -> bank-safe)

static __device__ __forceinline__ uint16_t f2bfbits(float f) {
  union { float f; uint32_t u; } c; c.f = f;
  uint32_t r = c.u + 0x7FFFu + ((c.u >> 16) & 1u);   // round-to-nearest-even
  return (uint16_t)(r >> 16);
}
static __device__ __forceinline__ float bfbits2f(uint16_t b) {
  union { uint32_t u; float f; } o; o.u = ((uint32_t)b) << 16;
  return o.f;
}

// ---------------------------------------------------------------------------
// Prep: codebook -> bf16 hi/lo split, plus exact fp32 ||e_k||^2.
// grid: 512 blocks (one code row) x 64 threads (4 floats each).
// ---------------------------------------------------------------------------
__global__ __launch_bounds__(64) void vq_prep(const float* __restrict__ cb,
                                              uint16_t* __restrict__ cb_hi,
                                              uint16_t* __restrict__ cb_lo,
                                              float* __restrict__ enorms) {
  __shared__ float sn[64];
  const int code = blockIdx.x;
  const int t = threadIdx.x;
  const float4 v = ((const float4*)(cb + (size_t)code * D_EMB))[t];
  const float x[4] = {v.x, v.y, v.z, v.w};
  ushort4 hv, lv;
  uint16_t hb[4], lb[4];
#pragma unroll
  for (int j = 0; j < 4; ++j) {
    hb[j] = f2bfbits(x[j]);
    lb[j] = f2bfbits(x[j] - bfbits2f(hb[j]));
  }
  hv.x = hb[0]; hv.y = hb[1]; hv.z = hb[2]; hv.w = hb[3];
  lv.x = lb[0]; lv.y = lb[1]; lv.z = lb[2]; lv.w = lb[3];
  ((ushort4*)(cb_hi + (size_t)code * D_EMB))[t] = hv;
  ((ushort4*)(cb_lo + (size_t)code * D_EMB))[t] = lv;

  sn[t] = x[0]*x[0] + x[1]*x[1] + x[2]*x[2] + x[3]*x[3];
  __syncthreads();
  for (int off = 32; off >= 1; off >>= 1) {
    if (t < off) sn[t] += sn[t + off];
    __syncthreads();
  }
  if (t == 0) enorms[code] = sn[0];
}

// ---------------------------------------------------------------------------
// Score + masked argmin. 8 waves x 16 rows per block; A tiles in VGPRs,
// B tiles double-buffered through LDS and shared by all waves.
// grid: 1024 blocks x 256 threads.
// ---------------------------------------------------------------------------
__global__ __launch_bounds__(256) void vq_scores(const int* __restrict__ node_type,
                                                 const float* __restrict__ z,
                                                 const uint16_t* __restrict__ cb_hi,
                                                 const uint16_t* __restrict__ cb_lo,
                                                 const float* __restrict__ enorms,
                                                 float* __restrict__ out_idx_f) {
  __shared__ uint16_t lds_bh[2][16 * BROW];   // 16.5 KB x2
  __shared__ uint16_t lds_bl[2][16 * BROW];
  __shared__ float    lds_en[NUM_EMB];        // 2 KB

  const int tid  = threadIdx.x;
  const int lane = tid & 31;
  const int wave = tid >> 5;
  const int m0   = (blockIdx.x * WAVES_PER_BLK + wave) * 16;
  const int h    = lane >> 4;      // K-half selector for A/B layouts
  const int mlo  = lane & 15;      // A: row index; B/C: column index

  // ---- stage ||e||^2 once ----
  lds_en[tid]       = enorms[tid];
  lds_en[tid + 256] = enorms[tid + 256];

  // ---- A tile: z[m0..m0+15][0..255] as hi/lo bf16 in ISA 16x32 layout ----
  // lane holds row (m0+mlo): elems 0..7  <- K = c*32 + 8h + 0..7
  //                          elems 8..15 <- K = c*32 + 16 + 8h + 0..7
  v16bf a_hi[8], a_lo[8];
  {
    const float* arow = z + (size_t)(m0 + mlo) * D_EMB;
#pragma unroll
    for (int c = 0; c < 8; ++c) {
      const float* p0 = arow + c * 32 + 8 * h;
      const float* p1 = p0 + 16;
      float x[16];
      const float4 f0 = ((const float4*)p0)[0];
      const float4 f1 = ((const float4*)p0)[1];
      const float4 f2 = ((const float4*)p1)[0];
      const float4 f3 = ((const float4*)p1)[1];
      x[0]=f0.x; x[1]=f0.y; x[2]=f0.z;  x[3]=f0.w;
      x[4]=f1.x; x[5]=f1.y; x[6]=f1.z;  x[7]=f1.w;
      x[8]=f2.x; x[9]=f2.y; x[10]=f2.z; x[11]=f2.w;
      x[12]=f3.x;x[13]=f3.y;x[14]=f3.z; x[15]=f3.w;
      union { v16bf v; uint16_t s[16]; } uh, ul;
#pragma unroll
      for (int e = 0; e < 16; ++e) {
        const uint16_t hb = f2bfbits(x[e]);
        uh.s[e] = hb;
        ul.s[e] = f2bfbits(x[e] - bfbits2f(hb));
      }
      a_hi[c] = uh.v;
      a_lo[c] = ul.v;
    }
  }

  // ---- group nibbles for this half's 8 rows ----
  uint32_t gbits = 0;
  {
    const int* ntp = node_type + m0 + 8 * h;
#pragma unroll
    for (int j = 0; j < 8; ++j) {
      const int nt = ntp[j];
      const uint32_t g = (nt == 5) ? 0u : (nt == 6) ? 1u : (nt == 7) ? 2u : 3u;
      gbits |= g << (4 * j);
    }
  }

  float best_s[8];
  int   best_i[8];
#pragma unroll
  for (int r = 0; r < 8; ++r) { best_s[r] = __builtin_inff(); best_i[r] = 0; }

  // ---- staging role of this thread: code-in-tile scl, 16-elem chunk sch ----
  const int scl = tid >> 4;
  const int sch = tid & 15;
  const size_t gsoff = (size_t)scl * D_EMB + (size_t)sch * 16;
  const int    lsoff = scl * BROW + sch * 16;

  // prologue: stage tile 0
  uint4 nh0, nh1, nl0, nl1;
  {
    const uint4* gh = (const uint4*)(cb_hi + gsoff);
    const uint4* gl = (const uint4*)(cb_lo + gsoff);
    nh0 = gh[0]; nh1 = gh[1]; nl0 = gl[0]; nl1 = gl[1];
  }
  *(uint4*)(&lds_bh[0][lsoff])     = nh0;
  *(uint4*)(&lds_bh[0][lsoff + 8]) = nh1;
  *(uint4*)(&lds_bl[0][lsoff])     = nl0;
  *(uint4*)(&lds_bl[0][lsoff + 8]) = nl1;
  __syncthreads();

  for (int t = 0; t < 32; ++t) {
    const int cur  = t & 1;
    const bool more = (t + 1) < 32;

    // issue global loads for tile t+1 before compute (latency hidden by WMMAs)
    if (more) {
      const size_t goff = (size_t)(t + 1) * (16 * D_EMB) + gsoff;
      const uint4* gh = (const uint4*)(cb_hi + goff);
      const uint4* gl = (const uint4*)(cb_lo + goff);
      nh0 = gh[0]; nh1 = gh[1]; nl0 = gl[0]; nl1 = gl[1];
      if (t + 2 < 32) __builtin_prefetch(cb_hi + goff + 16 * D_EMB, 0, 3);
    }

    // ---- compute tile t from LDS ----
    const int n = t * 16 + mlo;                 // code index for this lane
    const float en = lds_en[n];
    const uint16_t* bh = &lds_bh[cur][mlo * BROW + 8 * h];
    const uint16_t* bl = &lds_bl[cur][mlo * BROW + 8 * h];
    v8f acc0 = {};   // hi*hi
    v8f acc1 = {};   // hi*lo + lo*hi
#pragma unroll
    for (int c = 0; c < 8; ++c) {
      union { v16bf v; uint4 q[2]; } ubh, ubl;
      ubh.q[0] = *(const uint4*)(bh + c * 32);
      ubh.q[1] = *(const uint4*)(bh + c * 32 + 16);
      ubl.q[0] = *(const uint4*)(bl + c * 32);
      ubl.q[1] = *(const uint4*)(bl + c * 32 + 16);
      acc0 = __builtin_amdgcn_wmma_f32_16x16x32_bf16(false, a_hi[c], false, ubh.v,
                                                     (short)0, acc0, false, false);
      acc1 = __builtin_amdgcn_wmma_f32_16x16x32_bf16(false, a_lo[c], false, ubh.v,
                                                     (short)0, acc1, false, false);
      acc1 = __builtin_amdgcn_wmma_f32_16x16x32_bf16(false, a_hi[c], false, ubl.v,
                                                     (short)0, acc1, false, false);
    }

    const int colg = t >> 3;                    // quadrant of this column tile
#pragma unroll
    for (int r = 0; r < 8; ++r) {               // C layout: VGPR r -> row 8h+r
      const uint32_t g = (gbits >> (4 * r)) & 0xFu;
      const float s = (g == (uint32_t)colg) ? (en - 2.0f * (acc0[r] + acc1[r]))
                                            : __builtin_inff();
      if (s < best_s[r] || (s == best_s[r] && n < best_i[r])) {
        best_s[r] = s; best_i[r] = n;
      }
    }

    // ---- store tile t+1 into the other buffer ----
    if (more) {
      const int nb = (t + 1) & 1;
      *(uint4*)(&lds_bh[nb][lsoff])     = nh0;
      *(uint4*)(&lds_bh[nb][lsoff + 8]) = nh1;
      *(uint4*)(&lds_bl[nb][lsoff])     = nl0;
      *(uint4*)(&lds_bl[nb][lsoff + 8]) = nl1;
    }
    __syncthreads();
  }

  // ---- cross-lane argmin over the 16 columns of each half ----
#pragma unroll
  for (int r = 0; r < 8; ++r) {
    float s = best_s[r];
    int   i = best_i[r];
#pragma unroll
    for (int m = 8; m >= 1; m >>= 1) {
      const float so = __shfl_xor(s, m, 32);
      const int   io = __shfl_xor(i, m, 32);
      if (so < s || (so == s && io < i)) { s = so; i = io; }
    }
    if (mlo == 0) out_idx_f[m0 + 8 * h + r] = (float)i;
  }
}

// ---------------------------------------------------------------------------
// Gather quantized rows + deterministic per-block loss partials.
// grid: 2048 blocks x 256 threads; 64 rows per block, 64 elems per thread.
// out_q base is 8-byte aligned (offset 2+131072 floats) -> float2 stores.
// ---------------------------------------------------------------------------
__global__ __launch_bounds__(256) void vq_gather_loss(const float* __restrict__ z,
                                                      const float* __restrict__ cb,
                                                      const float* __restrict__ idx_f,
                                                      float* __restrict__ out_q,
                                                      float* __restrict__ partials) {
  __shared__ float sred[256];
  const int t = threadIdx.x;
  const int row = blockIdx.x * 64 + (t >> 2);
  const int q = t & 3;
  const int code = (int)idx_f[row];
  const float4* crow = (const float4*)(cb + (size_t)code * D_EMB) + q * 16;
  const float4* zrow = (const float4*)(z + (size_t)row * D_EMB) + q * 16;
  float2* orow = (float2*)(out_q + (size_t)row * D_EMB) + q * 32;
  float s = 0.0f;
#pragma unroll
  for (int j = 0; j < 16; ++j) {
    const float4 c4 = crow[j];
    const float4 z4 = zrow[j];
    const float dx = c4.x - z4.x, dy = c4.y - z4.y;
    const float dz = c4.z - z4.z, dw = c4.w - z4.w;
    s += dx * dx + dy * dy + dz * dz + dw * dw;
    float2 o0; o0.x = c4.x; o0.y = c4.y;
    float2 o1; o1.x = c4.z; o1.y = c4.w;
    orow[2 * j]     = o0;
    orow[2 * j + 1] = o1;
  }
  sred[t] = s;
  __syncthreads();
  for (int off = 128; off >= 1; off >>= 1) {
    if (t < off) sred[t] += sred[t + off];
    __syncthreads();
  }
  if (t == 0) partials[blockIdx.x] = sred[0];
}

// ---------------------------------------------------------------------------
// Final deterministic reduce of 2048 partials -> the two scalar losses.
// ---------------------------------------------------------------------------
__global__ __launch_bounds__(256) void vq_finalize(const float* __restrict__ partials,
                                                   float* __restrict__ out) {
  __shared__ double sred[256];
  const int t = threadIdx.x;
  double s = 0.0;
  for (int i = t; i < LOSS_BLOCKS; i += 256) s += (double)partials[i];
  sred[t] = s;
  __syncthreads();
  for (int off = 128; off >= 1; off >>= 1) {
    if (t < off) sred[t] += sred[t + off];
    __syncthreads();
  }
  if (t == 0) {
    const double mean = sred[0] / ((double)NROWS * (double)D_EMB);
    out[0] = (float)mean;          // vq_loss
    out[1] = (float)(0.25 * mean); // commitment_loss
  }
}

extern "C" void kernel_launch(void* const* d_in, const int* in_sizes, int n_in,
                              void* d_out, int out_size, void* d_ws, size_t ws_size,
                              hipStream_t stream) {
  const int*   node_type = (const int*)d_in[0];
  const float* z         = (const float*)d_in[1];
  const float* cb        = (const float*)d_in[2];
  float* out_f = (float*)d_out;

  char* ws = (char*)d_ws;
  uint16_t* cb_hi    = (uint16_t*)ws;                                   // 256 KB
  uint16_t* cb_lo    = (uint16_t*)(ws + (size_t)NUM_EMB * D_EMB * 2);   // 256 KB
  float*    enorms   = (float*)(ws + (size_t)NUM_EMB * D_EMB * 4);      // 2 KB
  float*    partials = (float*)(ws + (size_t)NUM_EMB * D_EMB * 4 + 4096); // 8 KB

  float* out_idx = out_f + 2;
  float* out_q   = out_f + 2 + NROWS;

  vq_prep<<<NUM_EMB, 64, 0, stream>>>(cb, cb_hi, cb_lo, enorms);
  vq_scores<<<SCORE_BLOCKS, 256, 0, stream>>>(node_type, z, cb_hi, cb_lo, enorms, out_idx);
  vq_gather_loss<<<LOSS_BLOCKS, 256, 0, stream>>>(z, cb, out_idx, out_q, partials);
  vq_finalize<<<1, 256, 0, stream>>>(partials, out_f);
}
